// GNN_UMLS_Model_27410481283439
// MI455X (gfx1250) — matrix operand order repaired and checked by reading, not compiled, not run on hardware
//
#include <hip/hip_runtime.h>
#include <hip/hip_bf16.h>

typedef __attribute__((ext_vector_type(16))) __bf16 v16bf;
typedef __attribute__((ext_vector_type(8)))  __bf16 v8bf;
typedef __attribute__((ext_vector_type(8)))  float  v8f;

#define LRELU(v) ((v) > 0.0f ? (v) : 0.2f * (v))

// ---------------------------------------------------------------------------
// f32 -> bf16 row-major conversion (activations)
// ---------------------------------------------------------------------------
__global__ void cvt_bf16(const float* __restrict__ src, __bf16* __restrict__ dst,
                         size_t total) {
  size_t i = (size_t)blockIdx.x * blockDim.x + threadIdx.x;
  if (i < total) dst[i] = (__bf16)src[i];
}

// ---------------------------------------------------------------------------
// W[Kin,Dout] f32 -> Wt[Dout,Kin] bf16 (LDS-tiled transpose, 32x32 tiles)
// block: 256 threads = 32x8
// ---------------------------------------------------------------------------
__global__ void wt_transpose_bf16(const float* __restrict__ W, __bf16* __restrict__ Wt,
                                  int Kin, int Dout) {
  __shared__ float tile[32][33];
  const int n0 = blockIdx.x * 32;   // Dout tile
  const int k0 = blockIdx.y * 32;   // Kin tile
  const int tx = threadIdx.x & 31;
  const int ty = threadIdx.x >> 5;  // 0..7
  #pragma unroll
  for (int i = 0; i < 32; i += 8)
    tile[ty + i][tx] = W[(size_t)(k0 + ty + i) * Dout + n0 + tx];
  __syncthreads();
  #pragma unroll
  for (int i = 0; i < 32; i += 8)
    Wt[(size_t)(n0 + ty + i) * Kin + k0 + tx] = (__bf16)tile[tx][ty + i];
}

// ---------------------------------------------------------------------------
// GEMM: C[N,Dout] = A[N,Kin] @ W[Kin,Dout], from bf16 A (row-major) and
// bf16 Wt = W^T (Dout x Kin, row-major), f32 accumulate via WMMA.
// One wave per block; 16x64 output tile; A fragment reused for 4 B fragments.
// Fragment layouts used here (A and B share one k order):
//   A 16x32 bf16: lane l -> M = l&15, K = 16*(l>>4) + {0..15}
//   B 32x16 bf16: lane l -> N = l&15, K = 16*(l>>4) + {0..15}
//   D 16x16 f32 : lane l -> N = l&15, VGPR r -> M = r + 8*(l>>4)
// ---------------------------------------------------------------------------
__global__ void __launch_bounds__(32)
gat_gemm_bf16(const __bf16* __restrict__ Abf, const __bf16* __restrict__ Wt,
              float* __restrict__ C, int Kin, int Dout) {
  const int lane = threadIdx.x;          // 0..31
  const int m0   = blockIdx.x * 16;      // row tile base
  const int n0   = blockIdx.y * 64;      // col tile base
  const int row  = lane & 15;
  const int kbA  = (lane >> 4) * 16;     // A K sub-offset: 0 or 16
  const int kbB  = (lane >> 4) * 16;     // B K sub-offset: 0 or 16

  v8f acc[4];
  #pragma unroll
  for (int t = 0; t < 4; ++t) acc[t] = (v8f){};

  const __bf16* arow = Abf + (size_t)(m0 + row) * Kin;

  for (int kk = 0; kk < Kin; kk += 32) {
    // A fragment: two contiguous 8-elem (16B) vector loads
    v8bf alo = *(const v8bf*)(arow + kk + kbA);
    v8bf ahi = *(const v8bf*)(arow + kk + kbA + 8);
    v16bf af;
    #pragma unroll
    for (int j = 0; j < 8; ++j) { af[j] = alo[j]; af[8 + j] = ahi[j]; }

    #pragma unroll
    for (int t = 0; t < 4; ++t) {
      // B fragment: one contiguous 16-elem (32B) run per lane from W^T
      const __bf16* bp = Wt + (size_t)(n0 + t * 16 + row) * Kin + kk + kbB;
      v16bf bf = *(const v16bf*)bp;
      acc[t] = __builtin_amdgcn_wmma_f32_16x16x32_bf16(
          false, af, false, bf, (short)0, acc[t], false, false);
    }
  }

  const int mhi = (lane >> 4) * 8;
  #pragma unroll
  for (int t = 0; t < 4; ++t) {
    #pragma unroll
    for (int r = 0; r < 8; ++r) {
      C[(size_t)(m0 + mhi + r) * Dout + n0 + t * 16 + row] = acc[t][r];
    }
  }
}

// ---------------------------------------------------------------------------
// Per-node attention scalars: ssrc[n] = h[n]·a_src, sdst[n] = h[n]·a_dst
// ---------------------------------------------------------------------------
__global__ void node_scalars(const float* __restrict__ h,
                             const float* __restrict__ as_,
                             const float* __restrict__ ad_,
                             int D, float* __restrict__ ssrc,
                             float* __restrict__ sdst) {
  const int n = blockIdx.x;
  const float* hp = h + (size_t)n * D;
  float s1 = 0.0f, s2 = 0.0f;
  for (int d = threadIdx.x; d < D; d += blockDim.x) {
    float v = hp[d];
    s1 += v * as_[d];
    s2 += v * ad_[d];
  }
  __shared__ float r1[256], r2[256];
  r1[threadIdx.x] = s1;
  r2[threadIdx.x] = s2;
  __syncthreads();
  for (int s = blockDim.x >> 1; s > 0; s >>= 1) {
    if ((int)threadIdx.x < s) {
      r1[threadIdx.x] += r1[threadIdx.x + s];
      r2[threadIdx.x] += r2[threadIdx.x + s];
    }
    __syncthreads();
  }
  if (threadIdx.x == 0) {
    ssrc[n] = r1[0];
    sdst[n] = r2[0];
  }
}

// ---------------------------------------------------------------------------
// Init: zero aggregation output, m = -inf, denom = 0
// ---------------------------------------------------------------------------
__global__ void init_aggr(float* __restrict__ out, size_t total,
                          float* __restrict__ m, float* __restrict__ denom, int N) {
  size_t i = (size_t)blockIdx.x * blockDim.x + threadIdx.x;
  if (i < total) out[i] = 0.0f;
  if (i < (size_t)N) {
    m[i] = __int_as_float(0xff800000);  // -inf
    denom[i] = 0.0f;
  }
}

// float atomic max via signed/unsigned integer atomics (valid for -inf init)
__device__ __forceinline__ void atomicMaxF(float* addr, float v) {
  if (v >= 0.0f)
    atomicMax((int*)addr, __float_as_int(v));
  else
    atomicMin((unsigned int*)addr, __float_as_uint(v));
}

// ---------------------------------------------------------------------------
// Edge pass 1: e = leaky_relu(ssrc[col] + sdst[row]); segment-max into m[row]
// ---------------------------------------------------------------------------
__global__ void edge_logits(const int* __restrict__ rows, const int* __restrict__ cols,
                            const float* __restrict__ ssrc, const float* __restrict__ sdst,
                            float* __restrict__ ebuf, float* __restrict__ m, int E) {
  int e = blockIdx.x * blockDim.x + threadIdx.x;
  if (e >= E) return;
  int r = rows[e], c = cols[e];
  float v = ssrc[c] + sdst[r];
  v = LRELU(v);
  ebuf[e] = v;
  atomicMaxF(&m[r], v);
}

// ---------------------------------------------------------------------------
// Edge pass 2: ex = exp(e - m[row]); denom[row] += ex
// ---------------------------------------------------------------------------
__global__ void edge_exp(const int* __restrict__ rows, float* __restrict__ ebuf,
                         const float* __restrict__ m, float* __restrict__ denom, int E) {
  int e = blockIdx.x * blockDim.x + threadIdx.x;
  if (e >= E) return;
  int r = rows[e];
  float ex = __expf(ebuf[e] - m[r]);
  ebuf[e] = ex;
  atomicAdd(&denom[r], ex);
}

// ---------------------------------------------------------------------------
// Edge pass 3: out[row,:] += (ex/denom[row]) * h[col,:]   (one block per edge)
// ---------------------------------------------------------------------------
__global__ void edge_scatter(const int* __restrict__ rows, const int* __restrict__ cols,
                             const float* __restrict__ ebuf, const float* __restrict__ denom,
                             const float* __restrict__ h, float* __restrict__ out, int D) {
  int e = blockIdx.x;
  int r = rows[e], c = cols[e];
  float alpha = ebuf[e] / denom[r];
  const float* hp = h + (size_t)c * D;
  float* op = out + (size_t)r * D;
  for (int d = threadIdx.x; d < D; d += blockDim.x)
    atomicAdd(&op[d], alpha * hp[d]);
}

// ---------------------------------------------------------------------------
// Bias (+ optional leaky relu) epilogue
// ---------------------------------------------------------------------------
__global__ void bias_act(float* __restrict__ x, const float* __restrict__ b,
                         size_t total, int D, int doRelu) {
  size_t i = (size_t)blockIdx.x * blockDim.x + threadIdx.x;
  if (i >= total) return;
  float v = x[i] + b[i % (size_t)D];
  if (doRelu) v = LRELU(v);
  x[i] = v;
}

// ---------------------------------------------------------------------------
// Final: out[i,j] = img_feat[i,:] · x3[mask[j],:]   (one wave per output elem)
// ---------------------------------------------------------------------------
__global__ void __launch_bounds__(32)
final_gemm(const float* __restrict__ img, const float* __restrict__ x3,
           const int* __restrict__ mask, float* __restrict__ out, int D, int NM) {
  int i = blockIdx.x;   // img row
  int j = blockIdx.y;   // mask slot
  const float* a = img + (size_t)i * D;
  const float* b = x3 + (size_t)mask[j] * D;
  float s = 0.0f;
  for (int d = threadIdx.x; d < D; d += 32) s += a[d] * b[d];
  #pragma unroll
  for (int off = 16; off > 0; off >>= 1) s += __shfl_down(s, off, 32);
  if (threadIdx.x == 0) out[(size_t)i * NM + j] = s;
}

// ---------------------------------------------------------------------------
// Host-side orchestration
// ---------------------------------------------------------------------------
static void run_layer(const float* x, int Din, const float* W, int Dout,
                      const float* as_, const float* ad_, const float* b,
                      int N, int E, const int* rows, const int* cols,
                      __bf16* xbf, __bf16* wt,
                      float* h, float* out, float* ssrc, float* sdst,
                      float* m, float* denom, float* ebuf, int doRelu,
                      hipStream_t stream) {
  // Pre-convert operands to bf16 (x row-major, W transposed)
  size_t xtotal = (size_t)N * Din;
  cvt_bf16<<<(unsigned)((xtotal + 255) / 256), 256, 0, stream>>>(x, xbf, xtotal);
  dim3 tGrid(Dout / 32, Din / 32);
  wt_transpose_bf16<<<tGrid, 256, 0, stream>>>(W, wt, Din, Dout);

  // h = x @ W  (bf16 WMMA, f32 accumulate)
  dim3 gGrid(N / 16, Dout / 64);
  gat_gemm_bf16<<<gGrid, 32, 0, stream>>>(xbf, wt, h, Din, Dout);

  // per-node scalars
  node_scalars<<<N, 256, 0, stream>>>(h, as_, ad_, Dout, ssrc, sdst);

  // init aggregation state
  size_t total = (size_t)N * Dout;
  init_aggr<<<(unsigned)((total + 255) / 256), 256, 0, stream>>>(out, total, m, denom, N);

  // segment softmax + scatter
  edge_logits<<<(E + 255) / 256, 256, 0, stream>>>(rows, cols, ssrc, sdst, ebuf, m, E);
  edge_exp<<<(E + 255) / 256, 256, 0, stream>>>(rows, ebuf, m, denom, E);
  edge_scatter<<<E, 256, 0, stream>>>(rows, cols, ebuf, denom, h, out, Dout);

  // bias + activation
  bias_act<<<(unsigned)((total + 255) / 256), 256, 0, stream>>>(out, b, total, Dout, doRelu);
}

extern "C" void kernel_launch(void* const* d_in, const int* in_sizes, int n_in,
                              void* d_out, int out_size, void* d_ws, size_t ws_size,
                              hipStream_t stream) {
  const float* node_features = (const float*)d_in[0];
  const float* img_feat      = (const float*)d_in[1];
  const int*   edge_index    = (const int*)d_in[2];
  const int*   mask_idx      = (const int*)d_in[3];
  const float* W1  = (const float*)d_in[4];
  const float* a1s = (const float*)d_in[5];
  const float* a1d = (const float*)d_in[6];
  const float* b1  = (const float*)d_in[7];
  const float* W2  = (const float*)d_in[8];
  const float* a2s = (const float*)d_in[9];
  const float* a2d = (const float*)d_in[10];
  const float* b2  = (const float*)d_in[11];
  const float* W3  = (const float*)d_in[12];
  const float* a3s = (const float*)d_in[13];
  const float* a3d = (const float*)d_in[14];
  const float* b3  = (const float*)d_in[15];

  const int D1 = in_sizes[5];             // 1024
  const int D2 = in_sizes[9];             // 1024
  const int D3 = in_sizes[13];            // 1664
  const int F0 = in_sizes[4] / D1;        // 768
  const int N  = in_sizes[0] / F0;        // 10000
  const int E  = in_sizes[2] / 2;         // 170000
  const int NM = in_sizes[3];             // 14
  const int MI = in_sizes[1] / D3;        // 32

  const int* rows = edge_index;           // edge_index[0]  (aggregation node)
  const int* cols = edge_index + E;       // edge_index[1]  (message source)

  const int DinMax = (D1 > F0) ? ((D2 > D1) ? D2 : D1) : ((D2 > F0) ? D2 : F0);

  // Workspace carving (256B aligned)
  auto align256 = [](size_t v) { return (v + 255) & ~(size_t)255; };
  const size_t big = align256((size_t)N * D3 * sizeof(float));
  char* w = (char*)d_ws;
  float*  P     = (float*)w;  w += big;                              // h buffer
  float*  Q     = (float*)w;  w += big;                              // x/out ping buffer
  __bf16* xbf   = (__bf16*)w; w += align256((size_t)N * DinMax * 2); // bf16 activations
  __bf16* wt    = (__bf16*)w; w += align256((size_t)D3 * DinMax * 2);// bf16 W^T
  float*  ssrc  = (float*)w;  w += align256((size_t)N * 4);
  float*  sdst  = (float*)w;  w += align256((size_t)N * 4);
  float*  m     = (float*)w;  w += align256((size_t)N * 4);
  float*  denom = (float*)w;  w += align256((size_t)N * 4);
  float*  ebuf  = (float*)w;  w += align256((size_t)E * 4);
  (void)ws_size; (void)n_in;

  // Layer 1: x = node_features -> Q
  run_layer(node_features, F0, W1, D1, a1s, a1d, b1, N, E, rows, cols,
            xbf, wt, P, Q, ssrc, sdst, m, denom, ebuf, /*relu=*/1, stream);
  // Layer 2: x = Q -> Q   (Q dead as GEMM input once xbf/h are produced)
  run_layer(Q, D1, W2, D2, a2s, a2d, b2, N, E, rows, cols,
            xbf, wt, P, Q, ssrc, sdst, m, denom, ebuf, /*relu=*/1, stream);
  // Layer 3: x = Q -> Q, no relu
  run_layer(Q, D2, W3, D3, a3s, a3d, b3, N, E, rows, cols,
            xbf, wt, P, Q, ssrc, sdst, m, denom, ebuf, /*relu=*/0, stream);

  // Final: out[32,14] = img_feat @ Q[mask].T
  dim3 fGrid(MI, NM);
  final_gemm<<<fGrid, 32, 0, stream>>>(img_feat, Q, mask_idx, (float*)d_out, D3, NM);
  (void)out_size;
}
